// ColoAttention_54133767798892
// MI455X (gfx1250) — compile-verified, matrix operand order checked by reading, not run on hardware
//
#include <hip/hip_runtime.h>

typedef __attribute__((ext_vector_type(16))) __bf16 v16bf;
typedef __attribute__((ext_vector_type(8)))  __bf16 v8bf;
typedef __attribute__((ext_vector_type(8)))  float  v8f;

#define B_  4
#define S_  2048
#define H_  16
#define D_  128
#define RS_ (H_ * D_)          // row stride in elements (2048)
#define NEGBIG (-1e30f)

static_assert(sizeof(__bf16) == 2, "bf16 size");

__device__ __forceinline__ v16bf ldsAB(const __bf16* lo, const __bf16* hi) {
  union { v16bf v; v8bf h[2]; } u;
  u.h[0] = *(const v8bf*)lo;
  u.h[1] = *(const v8bf*)hi;
  return u.v;
}

// DPP16 lane permute (stays within a 16-lane row; wave32 has two rows, which
// matches the C/D-layout halves). All-VALU: keeps the LDS pipe free for WMMA
// operand ds_load_b128s.
template <int CTRL>
__device__ __forceinline__ float dppf(float x) {
  int r = __builtin_amdgcn_update_dpp(0, __builtin_bit_cast(int, x),
                                      CTRL, 0xf, 0xf, true);
  return __builtin_bit_cast(float, r);
}

__device__ __forceinline__ float rowmax16(float x) {
  x = fmaxf(x, dppf<0xB1>(x));    // quad_perm [1,0,3,2]
  x = fmaxf(x, dppf<0x4E>(x));    // quad_perm [2,3,0,1]
  x = fmaxf(x, dppf<0x141>(x));   // row_half_mirror
  x = fmaxf(x, dppf<0x140>(x));   // row_mirror
  return x;
}

__device__ __forceinline__ float rowsum16(float x) {
  x += dppf<0xB1>(x);
  x += dppf<0x4E>(x);
  x += dppf<0x141>(x);
  x += dppf<0x140>(x);
  return x;
}

__device__ __forceinline__ unsigned pk2(float a, float b) {
  union { __bf16 h[2]; unsigned u; } z;
  z.h[0] = (__bf16)a; z.h[1] = (__bf16)b;
  return z.u;
}

__global__ __launch_bounds__(256, 1)
void flash_attn_bf16_wmma(const float* __restrict__ Q,
                          const float* __restrict__ K,
                          const float* __restrict__ V,
                          const int*  __restrict__ mask,
                          float* __restrict__ Out)
{
  // K tile row-major [64 keys][128 d], V tile transposed [128 d][64 keys],
  // P scratch per wave [16 q][64 k] -- all bf16.  48 KB total.
  __shared__ __align__(16) __bf16 Ks[64 * 128];
  __shared__ __align__(16) __bf16 Vt[128 * 64];
  __shared__ __align__(16) __bf16 Ps[8 * 16 * 64];

  const int b   = blockIdx.z;
  const int h   = blockIdx.y;
  const int tid = threadIdx.x;
  const int wave = tid >> 5;
  const int lane = tid & 31;
  const int half = lane >> 4;
  const int ln   = lane & 15;
  const int q0   = blockIdx.x * 128 + wave * 16;

  // ---- valid-prefix length for this batch: binary search (uniform) ----
  int lo = 0, hi = S_;
  while (lo < hi) {
    int mid = (lo + hi) >> 1;
    if (mask[b * S_ + mid] != 0) lo = mid + 1; else hi = mid;
  }
  const int len = lo;   // keys [0, len) valid

  const float scale = 0.08838834764831845f;  // 1/sqrt(128), folded into Q

  // ---- Load Q tile (16x128 fp32) -> 4 bf16 A-tiles (16x32), resident ----
  v16bf aQ[4];
  {
    const float* qrow = Q + ((size_t)b * S_ + q0 + ln) * RS_ + (size_t)h * D_;
#pragma unroll
    for (int dc = 0; dc < 4; ++dc) {
      const float* p0 = qrow + dc * 32 +      8 * half;
      const float* p1 = qrow + dc * 32 + 16 + 8 * half;
#pragma unroll
      for (int j = 0; j < 8; ++j) {
        aQ[dc][j]     = (__bf16)(p0[j] * scale);
        aQ[dc][8 + j] = (__bf16)(p1[j] * scale);
      }
    }
  }

  v8f oacc[8];
#pragma unroll
  for (int i = 0; i < 8; ++i) { v8f z = {}; oacc[i] = z; }
  float mrow[8], lrow[8];
#pragma unroll
  for (int r = 0; r < 8; ++r) { mrow[r] = NEGBIG; lrow[r] = 0.f; }

  __bf16* Pw = Ps + wave * 16 * 64;

  const size_t hdoff = (size_t)h * D_;

  for (int kt = 0; kt < len; kt += 64) {
    const size_t tilebase = ((size_t)b * S_ + kt) * RS_ + hdoff;

    // ---- Stage K: row-major bf16, contiguous 64B per thread -> b128s ----
    {
      const int row  = tid >> 2;     // key row 0..63
      const int quad = tid & 3;
      const float* kr = K + tilebase + (size_t)row * RS_;
#pragma unroll
      for (int j = 0; j < 8; ++j) {
        const int d0 = quad * 32 + j * 4;
        float4 kf = *(const float4*)(kr + d0);
        uint2 kp; kp.x = pk2(kf.x, kf.y); kp.y = pk2(kf.z, kf.w);
        *(uint2*)&Ks[row * 128 + d0] = kp;
      }
    }

    // ---- Stage V^T: 4x4 register-block transpose -> ds_store_b64 ----
    {
#pragma unroll
      for (int bb = 0; bb < 2; ++bb) {
        const int bi = tid * 2 + bb;        // 512 4x4 blocks
        const int kb = (bi >> 5) * 4;       // key base 0..60
        const int db = (bi & 31) * 4;       // d   base 0..124
        const float* vp = V + tilebase + db;
        float4 r0 = *(const float4*)(vp + (size_t)(kb + 0) * RS_);
        float4 r1 = *(const float4*)(vp + (size_t)(kb + 1) * RS_);
        float4 r2 = *(const float4*)(vp + (size_t)(kb + 2) * RS_);
        float4 r3 = *(const float4*)(vp + (size_t)(kb + 3) * RS_);
        uint2 t0; t0.x = pk2(r0.x, r1.x); t0.y = pk2(r2.x, r3.x);
        uint2 t1; t1.x = pk2(r0.y, r1.y); t1.y = pk2(r2.y, r3.y);
        uint2 t2; t2.x = pk2(r0.z, r1.z); t2.y = pk2(r2.z, r3.z);
        uint2 t3; t3.x = pk2(r0.w, r1.w); t3.y = pk2(r2.w, r3.w);
        *(uint2*)&Vt[(db + 0) * 64 + kb] = t0;
        *(uint2*)&Vt[(db + 1) * 64 + kb] = t1;
        *(uint2*)&Vt[(db + 2) * 64 + kb] = t2;
        *(uint2*)&Vt[(db + 3) * 64 + kb] = t3;
      }
    }
    __syncthreads();

    // ---- Prefetch next tile's K/V lines under this tile's compute ----
    if (kt + 64 < len) {
      const size_t nextbase = tilebase + (size_t)64 * RS_;
      const int row  = tid >> 2;
      const int quad = tid & 3;
      const char* kp = (const char*)(K + nextbase + (size_t)row * RS_) + quad * 128;
      const char* vp = (const char*)(V + nextbase + (size_t)row * RS_) + quad * 128;
      __builtin_prefetch(kp, 0, 0);   // global_prefetch_b8, 128B line each
      __builtin_prefetch(vp, 0, 0);
    }

    // per-lane key-padding bias (prefix mask vs len)
    const float mb0 = (kt + ln      < len) ? 0.f : NEGBIG;
    const float mb1 = (kt + 16 + ln < len) ? 0.f : NEGBIG;
    const float mb2 = (kt + 32 + ln < len) ? 0.f : NEGBIG;
    const float mb3 = (kt + 48 + ln < len) ? 0.f : NEGBIG;

    // ---- S = Q K^T : four 16-key subtiles x 4 chained WMMAs over D ----
    v8f s0 = {}, s1 = {}, s2 = {}, s3 = {};
#pragma unroll
    for (int dc = 0; dc < 4; ++dc) {
      const __bf16* k0 = &Ks[(ln     ) * 128 + dc * 32 + 8 * half];
      const __bf16* k1 = &Ks[(16 + ln) * 128 + dc * 32 + 8 * half];
      const __bf16* k2 = &Ks[(32 + ln) * 128 + dc * 32 + 8 * half];
      const __bf16* k3 = &Ks[(48 + ln) * 128 + dc * 32 + 8 * half];
      v16bf bk0 = ldsAB(k0, k0 + 16);
      v16bf bk1 = ldsAB(k1, k1 + 16);
      v16bf bk2 = ldsAB(k2, k2 + 16);
      v16bf bk3 = ldsAB(k3, k3 + 16);
      s0 = __builtin_amdgcn_wmma_f32_16x16x32_bf16(false, aQ[dc], false, bk0,
                                                   (short)0, s0, false, false);
      s1 = __builtin_amdgcn_wmma_f32_16x16x32_bf16(false, aQ[dc], false, bk1,
                                                   (short)0, s1, false, false);
      s2 = __builtin_amdgcn_wmma_f32_16x16x32_bf16(false, aQ[dc], false, bk2,
                                                   (short)0, s2, false, false);
      s3 = __builtin_amdgcn_wmma_f32_16x16x32_bf16(false, aQ[dc], false, bk3,
                                                   (short)0, s3, false, false);
    }

    // ---- online softmax over this 64-key tile; P -> LDS (bf16) ----
#pragma unroll
    for (int r = 0; r < 8; ++r) {
      float a0 = s0[r] + mb0;
      float a1 = s1[r] + mb1;
      float a2 = s2[r] + mb2;
      float a3 = s3[r] + mb3;
      float tmax = rowmax16(fmaxf(fmaxf(a0, a1), fmaxf(a2, a3)));
      float mnew  = fmaxf(mrow[r], tmax);
      float alpha = __expf(mrow[r] - mnew);
      float p0 = __expf(a0 - mnew);
      float p1 = __expf(a1 - mnew);
      float p2 = __expf(a2 - mnew);
      float p3 = __expf(a3 - mnew);
      float ts = rowsum16((p0 + p1) + (p2 + p3));
      lrow[r] = lrow[r] * alpha + ts;
      mrow[r] = mnew;
#pragma unroll
      for (int nc = 0; nc < 8; ++nc) oacc[nc][r] *= alpha;
      const int rr = r + 8 * half;               // C-layout row
      Pw[rr * 64 + ln]      = (__bf16)p0;
      Pw[rr * 64 + 16 + ln] = (__bf16)p1;
      Pw[rr * 64 + 32 + ln] = (__bf16)p2;
      Pw[rr * 64 + 48 + ln] = (__bf16)p3;
    }

    // in-wave LDS RAW: drain DS ops before re-reading P in A layout
    asm volatile("s_wait_dscnt 0" ::: "memory");

    // ---- O += P * V : A = P (16x64 -> 2 k-chunks), B = Vt (32x16) ----
    const __bf16* pa = &Pw[ln * 64 + 8 * half];
    v16bf aP0 = ldsAB(pa,      pa + 16);
    v16bf aP1 = ldsAB(pa + 32, pa + 48);
#pragma unroll
    for (int nc = 0; nc < 8; ++nc) {
      const __bf16* vb = &Vt[(nc * 16 + ln) * 64 + 8 * half];
      v16bf bv0 = ldsAB(vb,      vb + 16);
      v16bf bv1 = ldsAB(vb + 32, vb + 48);
      oacc[nc] = __builtin_amdgcn_wmma_f32_16x16x32_bf16(false, aP0, false, bv0,
                                                         (short)0, oacc[nc],
                                                         false, false);
      oacc[nc] = __builtin_amdgcn_wmma_f32_16x16x32_bf16(false, aP1, false, bv1,
                                                         (short)0, oacc[nc],
                                                         false, false);
    }

    __syncthreads();   // protect LDS tiles before next stage overwrites
  }

  // ---- epilogue: normalize, zero padded query rows, store fp32 ----
#pragma unroll
  for (int r = 0; r < 8; ++r) {
    const int row = q0 + r + 8 * half;
    const float qm  = (row < len) ? 1.f : 0.f;
    const float inv = qm / lrow[r];
    float* orow = Out + ((size_t)b * S_ + row) * RS_ + hdoff;
#pragma unroll
    for (int nc = 0; nc < 8; ++nc)
      orow[nc * 16 + ln] = oacc[nc][r] * inv;
  }
}

extern "C" void kernel_launch(void* const* d_in, const int* in_sizes, int n_in,
                              void* d_out, int out_size, void* d_ws, size_t ws_size,
                              hipStream_t stream) {
  (void)in_sizes; (void)n_in; (void)d_ws; (void)ws_size; (void)out_size;
  const float* q   = (const float*)d_in[0];
  const float* k   = (const float*)d_in[1];
  const float* v   = (const float*)d_in[2];
  const int*   msk = (const int*)d_in[3];
  float*       out = (float*)d_out;

  dim3 grid(S_ / 128, H_, B_);   // 128 query rows per WG (8 waves x 16 rows)
  dim3 block(256);
  flash_attn_bf16_wmma<<<grid, block, 0, stream>>>(q, k, v, msk, out);
}
